// SentenceDecoder_47244640256185
// MI455X (gfx1250) — compile-verified, matrix operand order checked by reading, not run on hardware
//
#include <hip/hip_runtime.h>

// LSTM sentence decoder for MI455X (gfx1250, wave32, WMMA bf16 16x16x32).
// Phases: (1) f32->bf16 conversion, (2) input-projection GEMM,
// (3) persistent 16-WG recurrent kernel: W_hh slice pinned in LDS via TDM
//     (tensor_load_to_lds + s_wait_tensorcnt), grid barrier per step,
// (4) output-projection GEMM (dominant 67 GFLOP; 2x4 register-blocked waves).

#define B_ 32
#define T_ 64
#define I_ 512
#define H_ 512
#define V_ 32000
#define G4_ (4 * H_)      // 2048 gate columns
#define NWG_REC 16

typedef __attribute__((ext_vector_type(16))) __bf16 v16bf;
typedef __attribute__((ext_vector_type(8)))  __bf16 v8bf;
typedef __attribute__((ext_vector_type(8)))  float  v8f;
typedef __attribute__((ext_vector_type(4)))  unsigned int u32x4;
typedef __attribute__((ext_vector_type(8)))  int i32x8;
typedef __attribute__((ext_vector_type(4)))  int i32x4;

#if defined(__has_builtin)
#  if __has_builtin(__builtin_amdgcn_tensor_load_to_lds)
#    define HAS_TDM 1
#  endif
#endif
#ifndef HAS_TDM
#  define HAS_TDM 0
#endif

union V16U { v16bf v; v8bf h[2]; };

__device__ __forceinline__ unsigned short f32_to_bf16_bits(float f) {
  union { float f; unsigned u; } x; x.f = f;
  unsigned r = x.u + 0x7FFFu + ((x.u >> 16) & 1u);   // round-to-nearest-even
  return (unsigned short)(r >> 16);
}

// One 16x32 bf16 WMMA fragment lane-slice.
// ISA 16-bit A layout: lanes 0-15 hold K = 0..7 and 16..23 (two 16B chunks),
// lanes 16-31 hold K = 8..15 and 24..31. `base` = row pointer + k-step offset.
__device__ __forceinline__ v16bf load_frag(const unsigned short* base, int half) {
  V16U u;
  u.h[0] = *reinterpret_cast<const v8bf*>(base + 8 * half);
  u.h[1] = *reinterpret_cast<const v8bf*>(base + 16 + 8 * half);
  return u.v;
}

__device__ __forceinline__ v8f wmma_bf16(v16bf a, v16bf b, v8f c) {
  return __builtin_amdgcn_wmma_f32_16x16x32_bf16(false, a, false, b,
                                                 (short)0, c, false, false);
}

__device__ __forceinline__ float sigmoidf_(float x) {
  return 1.0f / (1.0f + __expf(-x));
}
__device__ __forceinline__ float tanhf_(float x) {
  return 2.0f / (1.0f + __expf(-2.0f * x)) - 1.0f;
}

#if HAS_TDM
// TDM: DMA a contiguous run of `elems` bf16 (elems <= 65535, multiple of 8)
// from global memory into LDS.  D# per CDNA5 ISA 8.3/8.4:
//   group0: count=1, lds_addr, global_addr[56:0], type=2
//   group1: data_size=1 (2B), tensor_dim0=elems, tensor_dim1=1,
//           tile_dim0=elems, tile_dim1=1, tensor_dim0_stride=elems
__device__ __forceinline__ void tdm_load_1d(const unsigned short* gsrc,
                                            unsigned short* lds_dst,
                                            unsigned int elems) {
  unsigned long long ga = (unsigned long long)(uintptr_t)gsrc;
  unsigned int la = (unsigned int)(uintptr_t)lds_dst; // low 32 bits = LDS offset
  u32x4 g0;
  g0[0] = 1u;                                          // count=1
  g0[1] = la;                                          // lds_addr
  g0[2] = (unsigned int)(ga & 0xFFFFFFFFu);            // global_addr[31:0]
  g0[3] = (unsigned int)((ga >> 32) & 0x1FFFFFFu) | (2u << 30); // [56:32]|type=2
  i32x8 g1;
  g1[0] = (int)(1u << 16);                             // data_size=1 (2 bytes)
  g1[1] = (int)((elems & 0xFFFFu) << 16);              // tensor_dim0[15:0]
  g1[2] = (int)(((elems >> 16) & 0xFFFFu) | (1u << 16)); // dim0 hi | tensor_dim1=1
  g1[3] = (int)((elems & 0xFFFFu) << 16);              // tile_dim0 = elems
  g1[4] = 1;                                           // tile_dim1=1, tile_dim2=0
  g1[5] = (int)elems;                                  // tensor_dim0_stride lo
  g1[6] = 0;
  g1[7] = 0;
  i32x4 z4 = {0, 0, 0, 0};
#if __clang_major__ >= 23
  i32x8 z8 = {};
  __builtin_amdgcn_tensor_load_to_lds(g0, g1, z4, z4, z8, 0);
#else
  __builtin_amdgcn_tensor_load_to_lds(g0, g1, z4, z4, 0);
#endif
}
#endif

// ---------------------------------------------------------------- converts
__global__ void cvt_f32_bf16(const float* __restrict__ src,
                             unsigned short* __restrict__ dst, int n) {
  int i = blockIdx.x * blockDim.x + threadIdx.x;
  int stride = gridDim.x * blockDim.x;
  for (; i < n; i += stride) dst[i] = f32_to_bf16_bits(src[i]);
}

// isequence [B,T,I] -> time-major bf16 X [T*B, I]
__global__ void cvt_x_timemajor(const float* __restrict__ src,
                                unsigned short* __restrict__ dst) {
  int row = blockIdx.x;               // row = t*B + b
  int t = row / B_, b = row % B_;
  const float* s = src + ((size_t)b * T_ + t) * I_;
  unsigned short* d = dst + (size_t)row * I_;
  for (int i = threadIdx.x; i < I_; i += blockDim.x)
    d[i] = f32_to_bf16_bits(s[i]);
}

__global__ void zero_ctr(unsigned int* __restrict__ ctr, int n) {
  int i = threadIdx.x;
  if (i < n) ctr[i] = 0;
}

// ------------------------------------------------------------ generic GEMM
// C[m,n] = sum_k A[m,k]*W[n,k] + bias0[n] (+ bias1[n])
// mode 0: C row-major [M,N].  mode 1: m = t*B+b, store C[((b*T)+t)*N + n].
// Block = 256 thr = 8 waves arranged 2(m) x 4(n); block tile 64M x 256N.
// Each wave computes a 32x64 tile: 2 A-frags x 4 W-frags -> 8 WMMA per k-step
// (12 b128 loads per 8 WMMA instead of 4 per 1).
__global__ void wmma_gemm_bf16(const unsigned short* __restrict__ A,
                               const unsigned short* __restrict__ W,
                               const float* __restrict__ bias0,
                               const float* __restrict__ bias1,
                               float* __restrict__ C,
                               int M, int N, int K, int mode) {
  const int lane = threadIdx.x & 31;
  const int wave = threadIdx.x >> 5;     // 0..7
  const int wm = wave >> 2;              // 0..1
  const int wn = wave & 3;               // 0..3
  const int half = lane >> 4;
  const int r15 = lane & 15;

  const int m0 = blockIdx.x * 64 + wm * 32;
  const int n0 = blockIdx.y * 256 + wn * 64;

  const unsigned short* arow0 = A + (size_t)(m0 + r15) * K;
  const unsigned short* arow1 = A + (size_t)(m0 + 16 + r15) * K;
  const unsigned short* wrow  = W + (size_t)(n0 + r15) * K;

  v8f acc[2][4] = {};
  for (int k = 0; k < K; k += 32) {
    v16bf a0 = load_frag(arow0 + k, half);
    v16bf a1 = load_frag(arow1 + k, half);
#pragma unroll
    for (int nt = 0; nt < 4; ++nt) {
      const unsigned short* wr = wrow + (size_t)nt * 16 * K;
      if (k + 32 < K) __builtin_prefetch(wr + k + 32, 0, 1);
      v16bf bf = load_frag(wr + k, half);
      acc[0][nt] = wmma_bf16(a0, bf, acc[0][nt]);
      acc[1][nt] = wmma_bf16(a1, bf, acc[1][nt]);
    }
  }

#pragma unroll
  for (int nt = 0; nt < 4; ++nt) {
    int n = n0 + nt * 16 + r15;
    float bv = bias0[n] + (bias1 ? bias1[n] : 0.0f);
#pragma unroll
    for (int mt = 0; mt < 2; ++mt) {
#pragma unroll
      for (int r = 0; r < 8; ++r) {
        int m = m0 + mt * 16 + r + 8 * half;
        float v = acc[mt][nt][r] + bv;
        if (mode == 0) {
          C[(size_t)m * N + n] = v;
        } else {
          int tt = m >> 5, bb = m & 31;          // m = t*32 + b
          C[((size_t)bb * T_ + tt) * (size_t)N + n] = v;
        }
      }
    }
  }
}

// --------------------------------------------------------- recurrent kernel
// 16 persistent workgroups. WG g owns h-cols [g*32, g*32+32) and gate slices
// {q*512 + g*32 + jj}.  Its 128 W_hh rows (128 KB bf16, 4 contiguous 32 KB
// blocks) are DMA'd into LDS once via TDM and pinned for all 64 steps.
// h(t) (32 KB) is DMA'd into LDS each step.  Cell state c in registers.
__global__ void lstm_recurrent(const unsigned short* __restrict__ Whh, // [2048,512]
                               const float* __restrict__ xg,           // [T*B, 2048]
                               unsigned short* __restrict__ h_all,     // [(T+1)*B, 512]
                               unsigned int* __restrict__ ctr) {
  extern __shared__ __align__(16) char smem[];
  unsigned short* whh_s = (unsigned short*)smem;                        // 128*512
  unsigned short* h_s   = (unsigned short*)(smem + 128 * 512 * 2);      // 32*512
  float* gates_s        = (float*)(smem + 128 * 512 * 2 + 32 * 512 * 2);// 32*128

  const int tid = threadIdx.x;
  const int g = blockIdx.x;
  const int lane = tid & 31, wave = tid >> 5;
  const int half = lane >> 4, r15 = lane & 15;

  // Pin W_hh slice: local rows [q*32, q*32+32) = global rows [q*512+g*32, +32)
#if HAS_TDM
  if (wave == 0) {
#pragma unroll
    for (int q = 0; q < 4; ++q)
      tdm_load_1d(Whh + (size_t)(q * H_ + g * 32) * H_,
                  whh_s + q * 32 * H_, 32 * H_);
    __builtin_amdgcn_s_wait_tensorcnt(0);
  }
#else
  for (int idx = tid; idx < 128 * (H_ / 8); idx += blockDim.x) {
    int lg = idx / (H_ / 8), cc = idx % (H_ / 8);
    int grow = ((lg >> 5) * H_) + g * 32 + (lg & 31);
    *reinterpret_cast<uint4*>(whh_s + lg * H_ + cc * 8) =
        *reinterpret_cast<const uint4*>(Whh + (size_t)grow * H_ + cc * 8);
  }
#endif

  float c_reg[4] = {0.f, 0.f, 0.f, 0.f};
  const int jj = tid & 31;
  const int bbase = (tid >> 5) * 4;

  for (int t = 0; t < T_; ++t) {
    // stage h(t) (slot t of h_all; slot 0 = initial hidden) into LDS
#if HAS_TDM
    if (wave == 0) {
      tdm_load_1d(h_all + (size_t)t * B_ * H_, h_s, B_ * H_);
      __builtin_amdgcn_s_wait_tensorcnt(0);
    }
#else
    for (int idx = tid; idx < B_ * H_ / 8; idx += blockDim.x) {
      *reinterpret_cast<uint4*>(h_s + idx * 8) =
          *reinterpret_cast<const uint4*>(h_all + (size_t)t * B_ * H_ + idx * 8);
    }
#endif
    __syncthreads();

    // gates[:, my 128 cols] = h @ Whh_sliceT ; wave w: n-tile w, m-tiles 0,1
    v8f acc0 = {}, acc1 = {};
    const unsigned short* brow = whh_s + (wave * 16 + r15) * H_;
    const unsigned short* a0 = h_s + r15 * H_;
    const unsigned short* a1 = h_s + (16 + r15) * H_;
#pragma unroll
    for (int k = 0; k < H_; k += 32) {
      v16bf bf = load_frag(brow + k, half);
      acc0 = wmma_bf16(load_frag(a0 + k, half), bf, acc0);
      acc1 = wmma_bf16(load_frag(a1 + k, half), bf, acc1);
    }
    const int lg = wave * 16 + r15;
    const int ng = ((lg >> 5) * H_) + g * 32 + (lg & 31);
    const float* xgrow = xg + ((size_t)t * B_) * G4_ + ng;
#pragma unroll
    for (int r = 0; r < 8; ++r) {
      int b0 = r + 8 * half;
      gates_s[b0 * 128 + lg]        = acc0[r] + xgrow[(size_t)b0 * G4_];
      gates_s[(b0 + 16) * 128 + lg] = acc1[r] + xgrow[(size_t)(b0 + 16) * G4_];
    }
    __syncthreads();

    // activations + state update; write owned h-slice (bf16) to slot t+1
#pragma unroll
    for (int q = 0; q < 4; ++q) {
      int b = bbase + q;
      float ig = sigmoidf_(gates_s[b * 128 + jj]);
      float fg = sigmoidf_(gates_s[b * 128 + 32 + jj]);
      float gg = tanhf_(gates_s[b * 128 + 64 + jj]);
      float og = sigmoidf_(gates_s[b * 128 + 96 + jj]);
      float cn = fg * c_reg[q] + ig * gg;
      c_reg[q] = cn;
      float hn = og * tanhf_(cn);
      h_all[((size_t)(t + 1) * B_ + b) * H_ + g * 32 + jj] = f32_to_bf16_bits(hn);
    }

    // grid-wide barrier: all WGs must publish h(t+1) before anyone reads it
    __threadfence();
    __syncthreads();
    if (tid == 0) {
      atomicAdd(&ctr[t], 1u);
      while (atomicAdd(&ctr[t], 0u) < (unsigned)gridDim.x)
        __builtin_amdgcn_s_sleep(1);
    }
    __syncthreads();
  }
}

// ---------------------------------------------------------------- launcher
extern "C" void kernel_launch(void* const* d_in, const int* in_sizes, int n_in,
                              void* d_out, int out_size, void* d_ws, size_t ws_size,
                              hipStream_t stream) {
  const float* iseq = (const float*)d_in[0];
  const float* h0   = (const float*)d_in[1];
  const float* Wih  = (const float*)d_in[2];
  const float* Whh  = (const float*)d_in[3];
  const float* bih  = (const float*)d_in[4];
  const float* bhh  = (const float*)d_in[5];
  const float* Wout = (const float*)d_in[6];
  const float* bout = (const float*)d_in[7];
  float* out = (float*)d_out;

  // workspace layout (bytes), 256B-aligned segments; total ~58 MB
  constexpr size_t OF_CTR  = 0;
  constexpr size_t OF_WIH  = 1024;
  constexpr size_t OF_WHH  = OF_WIH  + (size_t)G4_ * I_ * 2;
  constexpr size_t OF_WOUT = OF_WHH  + (size_t)G4_ * H_ * 2;
  constexpr size_t OF_X    = OF_WOUT + (size_t)V_  * H_ * 2;
  constexpr size_t OF_XG   = OF_X    + (size_t)T_ * B_ * I_ * 2;
  constexpr size_t OF_HALL = OF_XG   + (size_t)T_ * B_ * G4_ * 4;

  char* ws = (char*)d_ws;
  unsigned int*   ctr    = (unsigned int*)(ws + OF_CTR);
  unsigned short* wih_b  = (unsigned short*)(ws + OF_WIH);
  unsigned short* whh_b  = (unsigned short*)(ws + OF_WHH);
  unsigned short* wout_b = (unsigned short*)(ws + OF_WOUT);
  unsigned short* x_b    = (unsigned short*)(ws + OF_X);
  float*          xgf    = (float*)(ws + OF_XG);
  unsigned short* hall_b = (unsigned short*)(ws + OF_HALL); // (T+1)*B*H, slot0=h0

  zero_ctr<<<1, 64, 0, stream>>>(ctr, T_);

  cvt_f32_bf16<<<256, 256, 0, stream>>>(Wih,  wih_b,  G4_ * I_);
  cvt_f32_bf16<<<256, 256, 0, stream>>>(Whh,  whh_b,  G4_ * H_);
  cvt_f32_bf16<<<1024, 256, 0, stream>>>(Wout, wout_b, V_ * H_);
  cvt_f32_bf16<<<32, 256, 0, stream>>>(h0,   hall_b, B_ * H_);
  cvt_x_timemajor<<<T_ * B_, 256, 0, stream>>>(iseq, x_b);

  // x_gates[t*B+b, g] = X·W_ihT + b_ih + b_hh   (M=2048, N=2048, K=512)
  wmma_gemm_bf16<<<dim3(T_ * B_ / 64, G4_ / 256), 256, 0, stream>>>(
      x_b, wih_b, bih, bhh, xgf, T_ * B_, G4_, I_, 0);

  // recurrence: 16 WGs, 176 KB dynamic LDS each (W_hh slice + h + gates)
  constexpr int LDS_BYTES = 128 * 512 * 2 + 32 * 512 * 2 + 32 * 128 * 4;
  (void)hipFuncSetAttribute((const void*)lstm_recurrent,
                            hipFuncAttributeMaxDynamicSharedMemorySize, LDS_BYTES);
  lstm_recurrent<<<NWG_REC, 256, LDS_BYTES, stream>>>(whh_b, xgf, hall_b, ctr);

  // logits: h_all(slots 1..T) · W_outT + b_out  (M=2048, N=32000, K=512)
  wmma_gemm_bf16<<<dim3(T_ * B_ / 64, V_ / 256), 256, 0, stream>>>(
      hall_b + (size_t)B_ * H_, wout_b, bout, nullptr, out, T_ * B_, V_, H_, 1);
}
